// DeconvCNNLoss_39934605918781
// MI455X (gfx1250) — compile-verified, matrix operand order checked by reading, not run on hardware
//
#include <hip/hip_runtime.h>

typedef __attribute__((ext_vector_type(16))) __bf16 v16bf;
typedef __attribute__((ext_vector_type(8)))  float  v8f;

#define TAU_INV 0.1f
#define B_ 4
#define E_ 512
#define L_ 512
#define V_ 32000

#define LTILE 64              // L rows per block (4 sub-tiles of 16)
#define ROWP  520             // ushorts per LDS row (512 + 8 pad, keeps 16B align)
#define VSPLIT 40             // V chunks -> grid.x
#define VCHUNK (V_ / VSPLIT)  // 800 (50 v-tiles of 16)

union Frag {
    v16bf v;
    unsigned int u[8];
    uint4 q[2];
};

// ---------------- K1: up numerator + zero the down accumulator ----------------
__global__ void up_init_kernel(const float* __restrict__ EN,
                               const float* __restrict__ DE,
                               float* __restrict__ up,
                               float* __restrict__ down) {
    int idx = blockIdx.x * blockDim.x + threadIdx.x;   // 0 .. B*L-1
    if (idx >= B_ * L_) return;
    int b = idx / L_;
    int l = idx - b * L_;
    const float* en = EN + (size_t)b * E_ * L_ + l;
    const float* de = DE + (size_t)b * E_ * L_ + l;
    float s = 0.f;
#pragma unroll 8
    for (int e = 0; e < E_; ++e) s += en[(size_t)e * L_] * de[(size_t)e * L_];
    up[idx]   = __expf(TAU_INV * s);
    down[idx] = 0.f;
}

// ---------------- K2: fused split-bf16 WMMA GEMM + exp + sum_v ----------------
__launch_bounds__(256, 1)
__global__ void gemm_exp_kernel(const float* __restrict__ DE,
                                const float* __restrict__ M,
                                float* __restrict__ down) {
    extern __shared__ unsigned short smem[];           // hi plane then lo plane
    unsigned short* aHi = smem;                        // [LTILE][ROWP]
    unsigned short* aLo = smem + LTILE * ROWP;

    const int b       = blockIdx.z;
    const int l0      = blockIdx.y * LTILE;
    const int v_begin = blockIdx.x * VCHUNK;

    // ---- stage A tile: DE[b, e, l0+r] -> bf16 hi/lo planes, layout [row][k]
    {
        const int r  = threadIdx.x & 63;               // l-row within tile
        const int e0 = threadIdx.x >> 6;               // 0..3
        const float* src = DE + (size_t)b * (E_ * L_) + l0 + r;
        for (int e = e0; e < E_; e += 4) {
            float x = src[(size_t)e * L_];
            unsigned int xb = __float_as_uint(x);
            unsigned int hi = xb & 0xFFFF0000u;        // truncated bf16 (hi part)
            float lo = x - __uint_as_float(hi);        // exact residual
            aHi[r * ROWP + e] = (unsigned short)(xb >> 16);
            aLo[r * ROWP + e] = (unsigned short)(__float_as_uint(lo) >> 16);
        }
    }
    __syncthreads();

    const int lane = threadIdx.x & 31;
    const int wave = threadIdx.x >> 5;                 // 0..7
    const int hl   = lane >> 4;                        // half: 0 or 1
    const int lid  = lane & 15;

    v8f esum[4];
#pragma unroll
    for (int s = 0; s < 4; ++s) esum[s] = (v8f){0,0,0,0,0,0,0,0};

    const int numTiles = VCHUNK / 16;                  // 50
    for (int t = wave; t < numTiles; t += 8) {
        const int v0 = v_begin + t * 16;
        // this lane supplies B column n = lid -> M row (v0+lid), contiguous in e.
        // Pointer pre-biased by this half's K stripe (hl*8).
        const float* mrow = M + (size_t)(v0 + lid) * E_ + hl * 8;

        v8f acc[4];
#pragma unroll
        for (int s = 0; s < 4; ++s) acc[s] = (v8f){0,0,0,0,0,0,0,0};

        // B software pipeline: preload K-step 0
        float4 g0 = *(const float4*)(mrow);
        float4 g1 = *(const float4*)(mrow + 4);
        float4 g2 = *(const float4*)(mrow + 16);
        float4 g3 = *(const float4*)(mrow + 20);

        for (int k0 = 0; k0 < E_; k0 += 32) {
            // consume current B regs; immediately issue next K-step's loads
            float4 c0 = g0, c1 = g1, c2 = g2, c3 = g3;
            if (k0 + 32 < E_) {
                const float* nm = mrow + k0 + 32;
                g0 = *(const float4*)(nm);
                g1 = *(const float4*)(nm + 4);
                g2 = *(const float4*)(nm + 16);
                g3 = *(const float4*)(nm + 20);
            }

            // ---- issue ALL 16 A-fragment ds_load_b128 up front
            const int koff = k0 + hl * 8;
            Frag ahi[4], alo[4];
#pragma unroll
            for (int s = 0; s < 4; ++s) {
                const unsigned short* rh = aHi + (16*s + lid) * ROWP + koff;
                const unsigned short* rl = aLo + (16*s + lid) * ROWP + koff;
                ahi[s].q[0] = *(const uint4*)(rh);
                ahi[s].q[1] = *(const uint4*)(rh + 16);
                alo[s].q[0] = *(const uint4*)(rl);
                alo[s].q[1] = *(const uint4*)(rl + 16);
            }

            // ---- convert B to bf16 hi/lo fragments (split precision)
            Frag bhi, blo;
            float cs[16] = {c0.x,c0.y,c0.z,c0.w, c1.x,c1.y,c1.z,c1.w,
                            c2.x,c2.y,c2.z,c2.w, c3.x,c3.y,c3.z,c3.w};
#pragma unroll
            for (int i = 0; i < 8; ++i) {
                unsigned int ua = __float_as_uint(cs[2*i]);
                unsigned int ub = __float_as_uint(cs[2*i+1]);
                unsigned int ha = ua & 0xFFFF0000u;
                unsigned int hb = ub & 0xFFFF0000u;
                bhi.u[i] = hb | (ha >> 16);
                float la = cs[2*i]   - __uint_as_float(ha);
                float lb = cs[2*i+1] - __uint_as_float(hb);
                blo.u[i] = (__float_as_uint(lb) & 0xFFFF0000u)
                         | (__float_as_uint(la) >> 16);
            }

            // ---- 12 WMMAs: 4 independent accumulator chains x 3 split terms
#pragma unroll
            for (int s = 0; s < 4; ++s) {
                acc[s] = __builtin_amdgcn_wmma_f32_16x16x32_bf16(
                            false, ahi[s].v, false, bhi.v, (short)0, acc[s], false, false);
                acc[s] = __builtin_amdgcn_wmma_f32_16x16x32_bf16(
                            false, alo[s].v, false, bhi.v, (short)0, acc[s], false, false);
                acc[s] = __builtin_amdgcn_wmma_f32_16x16x32_bf16(
                            false, ahi[s].v, false, blo.v, (short)0, acc[s], false, false);
            }
        }

        // ---- epilogue for this v-tile: exp and accumulate along v (lane dim)
#pragma unroll
        for (int s = 0; s < 4; ++s)
#pragma unroll
            for (int i = 0; i < 8; ++i)
                esum[s][i] += __expf(TAU_INV * acc[s][i]);
    }

    // ---- reduce across the 16 lanes of each half (N dimension), then atomics.
    // D layout: VGPR i / lanes 0-15 -> row 16s+i ; lanes 16-31 -> row 16s+8+i.
#pragma unroll
    for (int s = 0; s < 4; ++s) {
#pragma unroll
        for (int i = 0; i < 8; ++i) {
            float v = esum[s][i];
            v += __shfl_xor(v, 1, 32);
            v += __shfl_xor(v, 2, 32);
            v += __shfl_xor(v, 4, 32);
            v += __shfl_xor(v, 8, 32);
            if (lid == 0) {
                int row = l0 + 16*s + i + hl*8;
                atomicAdd(&down[b * L_ + row], v);
            }
        }
    }
}

// ---------------- K3: final scalar reduction ----------------
__global__ void finish_kernel(const float* __restrict__ up,
                              const float* __restrict__ down,
                              float* __restrict__ out) {
    __shared__ float red[256];
    float s = 0.f;
    for (int i = threadIdx.x; i < B_ * L_; i += 256) s += up[i] / down[i];
    red[threadIdx.x] = s;
    __syncthreads();
    for (int w = 128; w > 0; w >>= 1) {
        if (threadIdx.x < w) red[threadIdx.x] += red[threadIdx.x + w];
        __syncthreads();
    }
    if (threadIdx.x == 0) out[0] = red[0];
}

extern "C" void kernel_launch(void* const* d_in, const int* in_sizes, int n_in,
                              void* d_out, int out_size, void* d_ws, size_t ws_size,
                              hipStream_t stream) {
    (void)in_sizes; (void)n_in; (void)out_size; (void)ws_size;
    const float* EN = (const float*)d_in[0];   // embed_EN [B,E,L]
    const float* DE = (const float*)d_in[1];   // embed_DE [B,E,L]
    const float* M  = (const float*)d_in[2];   // embed_M  [V,E]
    float* up   = (float*)d_ws;                // B*L floats
    float* down = up + B_ * L_;                // B*L floats
    float* out  = (float*)d_out;

    up_init_kernel<<<(B_ * L_ + 255) / 256, 256, 0, stream>>>(EN, DE, up, down);

    dim3 grid(VSPLIT, L_ / LTILE, B_);         // 40 x 8 x 4 = 1280 blocks
    size_t smem = (size_t)2 * LTILE * ROWP * sizeof(unsigned short); // ~133 KB (<320 KB WGP LDS)
    gemm_exp_kernel<<<grid, 256, smem, stream>>>(DE, M, down);

    finish_kernel<<<1, 256, 0, stream>>>(up, down, out);
}
